// quantLinear_21139829031527
// MI455X (gfx1250) — 8-bit matrix operand order repaired and checked by reading, compiled, not run on hardware
//
#include <hip/hip_runtime.h>


typedef __attribute__((ext_vector_type(8))) int v8i;

#define K_DIM  2048
#define M_ROWS 16384
#define N_ROWS 2048
#define KC     128
#define LDK    144

union U8frag {
  v8i v;
  uint4 u4[2];
  unsigned long long q[4];
};

__device__ __forceinline__ int imax(int a, int b) { return a > b ? a : b; }
__device__ __forceinline__ int imin(int a, int b) { return a < b ? a : b; }

__device__ __forceinline__ void async_load_b128(unsigned lds_off,
                                                const void* gptr) {
  asm volatile("global_load_async_to_lds_b128 %0, %1, off"
               :: "v"(lds_off), "v"(gptr)
               : "memory");
}

__device__ __forceinline__ void wait_async0() {
  asm volatile("s_wait_asynccnt 0" ::: "memory");
}

__global__ __launch_bounds__(256)
void quant_x_kernel(const float* __restrict__ x,
                    unsigned char* __restrict__ xq,
                    float4* __restrict__ pX) {
  const int row  = blockIdx.x;
  const int tid  = threadIdx.x;
  const int lane = tid & 31;
  const int wave = tid >> 5;
  const float* xr = x + (size_t)row * K_DIM + tid * 8;

  float v[8];
  float4 f0 = *(const float4*)(xr);
  float4 f1 = *(const float4*)(xr + 4);
  v[0] = f0.x; v[1] = f0.y; v[2] = f0.z; v[3] = f0.w;
  v[4] = f1.x; v[5] = f1.y; v[6] = f1.z; v[7] = f1.w;

  float mx = v[0], mn = v[0];
  for (int j = 1; j < 8; ++j) { mx = fmaxf(mx, v[j]); mn = fminf(mn, v[j]); }
  for (int off = 16; off >= 1; off >>= 1) {
    mx = fmaxf(mx, __shfl_xor(mx, off, 32));
    mn = fminf(mn, __shfl_xor(mn, off, 32));
  }
  __shared__ float smax[8], smin[8];
  __shared__ int   ssum[8];
  if (lane == 0) { smax[wave] = mx; smin[wave] = mn; }
  __syncthreads();
  mx = smax[0]; mn = smin[0];
  for (int w = 1; w < 8; ++w) { mx = fmaxf(mx, smax[w]); mn = fminf(mn, smin[w]); }

  const float scale = fmaxf(mx - mn, 1e-5f) * (1.0f / 255.0f);
  const float zero  = fminf(fmaxf(rintf(-mn / scale), 0.0f), 255.0f);

  int wi[8];
  for (int j = 0; j < 8; ++j)
    wi[j] = (int)fminf(fmaxf(rintf(v[j] / scale) + zero, 0.0f), 255.0f);

  int gmx = wi[0], gmn = wi[0];
  for (int j = 1; j < 8; ++j) { gmx = imax(gmx, wi[j]); gmn = imin(gmn, wi[j]); }
  for (int off = 1; off <= 8; off <<= 1) {
    gmx = imax(gmx, __shfl_xor(gmx, off, 32));
    gmn = imin(gmn, __shfl_xor(gmn, off, 32));
  }
  const int range = gmx - gmn;
  const int even  = (gmx + gmn) >> 1;
  if (range <= 63 && range > 15)
    for (int j = 0; j < 8; ++j)
      wi[j] = (((wi[j] - even) >> 2) << 2) + even;

  unsigned int lo = (unsigned)wi[0] | ((unsigned)wi[1] << 8) |
                    ((unsigned)wi[2] << 16) | ((unsigned)wi[3] << 24);
  unsigned int hi = (unsigned)wi[4] | ((unsigned)wi[5] << 8) |
                    ((unsigned)wi[6] << 16) | ((unsigned)wi[7] << 24);
  *(uint2*)(xq + (size_t)row * K_DIM + tid * 8) = make_uint2(lo, hi);

  int s = 0;
  for (int j = 0; j < 8; ++j) s += wi[j];
  for (int off = 16; off >= 1; off >>= 1) s += __shfl_xor(s, off, 32);
  if (lane == 0) ssum[wave] = s;
  __syncthreads();
  if (tid == 0) {
    int tot = 0;
    for (int w = 0; w < 8; ++w) tot += ssum[w];
    pX[row] = make_float4(scale, zero, (float)tot, 0.0f);
  }
}

__global__ __launch_bounds__(256)
void quant_w_kernel(const float* __restrict__ wgt,
                    const float* __restrict__ bias,
                    unsigned char* __restrict__ wq,
                    float4* __restrict__ pW) {
  const int row  = blockIdx.x;
  const int tid  = threadIdx.x;
  const int lane = tid & 31;
  const int wave = tid >> 5;
  const float* wr = wgt + (size_t)row * K_DIM + tid * 8;

  float v[8];
  float4 f0 = *(const float4*)(wr);
  float4 f1 = *(const float4*)(wr + 4);
  v[0] = f0.x; v[1] = f0.y; v[2] = f0.z; v[3] = f0.w;
  v[4] = f1.x; v[5] = f1.y; v[6] = f1.z; v[7] = f1.w;

  float mx = v[0], mn = v[0];
  for (int j = 1; j < 8; ++j) { mx = fmaxf(mx, v[j]); mn = fminf(mn, v[j]); }
  for (int off = 16; off >= 1; off >>= 1) {
    mx = fmaxf(mx, __shfl_xor(mx, off, 32));
    mn = fminf(mn, __shfl_xor(mn, off, 32));
  }
  __shared__ float smax[8], smin[8];
  __shared__ int   ssum[8];
  if (lane == 0) { smax[wave] = mx; smin[wave] = mn; }
  __syncthreads();
  mx = smax[0]; mn = smin[0];
  for (int w = 1; w < 8; ++w) { mx = fmaxf(mx, smax[w]); mn = fminf(mn, smin[w]); }

  const float scale = fmaxf(mx - mn, 1e-5f) * (1.0f / 15.0f);
  const float zero  = fminf(fmaxf(rintf(-mn / scale), 0.0f), 15.0f);

  int wi[8];
  int s = 0;
  for (int j = 0; j < 8; ++j) {
    wi[j] = (int)fminf(fmaxf(rintf(v[j] / scale) + zero, 0.0f), 15.0f);
    s += wi[j];
  }
  unsigned int lo = (unsigned)wi[0] | ((unsigned)wi[1] << 8) |
                    ((unsigned)wi[2] << 16) | ((unsigned)wi[3] << 24);
  unsigned int hi = (unsigned)wi[4] | ((unsigned)wi[5] << 8) |
                    ((unsigned)wi[6] << 16) | ((unsigned)wi[7] << 24);
  *(uint2*)(wq + (size_t)row * K_DIM + tid * 8) = make_uint2(lo, hi);

  for (int off = 16; off >= 1; off >>= 1) s += __shfl_xor(s, off, 32);
  if (lane == 0) ssum[wave] = s;
  __syncthreads();
  if (tid == 0) {
    int tot = 0;
    for (int w = 0; w < 8; ++w) tot += ssum[w];
    pW[row] = make_float4(scale, zero, (float)tot, bias[row]);
  }
}

__global__ __launch_bounds__(256)
void wmma_gemm_kernel(const unsigned char* __restrict__ xq,
                      const unsigned char* __restrict__ wq,
                      const float4* __restrict__ pX,
                      const float4* __restrict__ pW,
                      float* __restrict__ out) {
  __shared__ __attribute__((aligned(16))) unsigned char sA[2][128 * LDK];
  __shared__ __attribute__((aligned(16))) unsigned char sB[2][128 * LDK];

  const int tid   = threadIdx.x;
  const int lane  = tid & 31;
  const int wave  = tid >> 5;
  const int waveM = wave >> 2;
  const int waveN = wave & 3;
  const int mBase = blockIdx.y * 128;
  const int nBase = blockIdx.x * 128;

  const int ldRow = tid >> 3;
  const int ldCol = (tid & 7) * 16;

  const int laneLo  = lane & 15;
  const int laneHiA = (lane >> 4) << 3;
  const int laneHiB = (lane >> 4) << 3;

  v8i acc[4][2] = {};

  for (int i = 0; i < 4; ++i) {
    const int r = ldRow + i * 32;
    async_load_b128((unsigned)(size_t)&sA[0][r * LDK + ldCol],
                    xq + (size_t)(mBase + r) * K_DIM + ldCol);
    async_load_b128((unsigned)(size_t)&sB[0][r * LDK + ldCol],
                    wq + (size_t)(nBase + r) * K_DIM + ldCol);
  }
  wait_async0();
  __syncthreads();

  for (int kc = 0; kc < K_DIM; kc += KC) {
    const int cur = (kc / KC) & 1;
    const int nxt = cur ^ 1;

    if (kc + KC < K_DIM) {
      for (int i = 0; i < 4; ++i) {
        const int r = ldRow + i * 32;
        async_load_b128((unsigned)(size_t)&sA[nxt][r * LDK + ldCol],
                        xq + (size_t)(mBase + r) * K_DIM + kc + KC + ldCol);
        async_load_b128((unsigned)(size_t)&sB[nxt][r * LDK + ldCol],
                        wq + (size_t)(nBase + r) * K_DIM + kc + KC + ldCol);
      }
    }

    for (int ks = 0; ks < KC; ks += 64) {
      U8frag b[2];
      for (int nj = 0; nj < 2; ++nj) {
        const unsigned char* pb =
            &sB[cur][(waveN * 32 + nj * 16 + laneLo) * LDK + ks + laneHiB];
        b[nj].q[0] = *(const unsigned long long*)(pb);
        b[nj].q[1] = *(const unsigned long long*)(pb + 16);
        b[nj].q[2] = *(const unsigned long long*)(pb + 32);
        b[nj].q[3] = *(const unsigned long long*)(pb + 48);
      }
      for (int mi = 0; mi < 4; ++mi) {
        U8frag a;
        const unsigned char* pa =
            &sA[cur][(waveM * 64 + mi * 16 + laneLo) * LDK + ks + laneHiA];
        a.q[0] = *(const unsigned long long*)(pa);
        a.q[1] = *(const unsigned long long*)(pa + 16);
        a.q[2] = *(const unsigned long long*)(pa + 32);
        a.q[3] = *(const unsigned long long*)(pa + 48);
        for (int nj = 0; nj < 2; ++nj) {
          acc[mi][nj] = __builtin_amdgcn_wmma_i32_16x16x64_iu8(
               false, a.v,  false, b[nj].v,
              acc[mi][nj],  false,  false);
        }
      }
    }

    wait_async0();
    __syncthreads();
  }

  for (int mi = 0; mi < 4; ++mi) {
    const int mFrag = mBase + waveM * 64 + mi * 16 + ((lane >> 4) << 3);
    float4 px[8];
    for (int r = 0; r < 8; ++r) px[r] = pX[mFrag + r];
    for (int nj = 0; nj < 2; ++nj) {
      const int n = nBase + waveN * 32 + nj * 16 + laneLo;
      const float4 pw = pW[n];
      for (int r = 0; r < 8; ++r) {
        const float accf = (float)acc[mi][nj][r];
        const float corr = accf - pw.y * px[r].z - px[r].y * pw.z
                         + (float)K_DIM * px[r].y * pw.y;
        out[(size_t)(mFrag + r) * N_ROWS + n] = px[r].x * pw.x * corr + pw.w;
      }
    }
  }
}

extern "C" void kernel_launch(void* const* d_in, const int* in_sizes, int n_in,
                              void* d_out, int out_size, void* d_ws, size_t ws_size,
                              hipStream_t stream) {
  const float* x    = (const float*)d_in[0];
  const float* wgt  = (const float*)d_in[1];
  const float* bias = (const float*)d_in[2];
  float* out = (float*)d_out;

  unsigned char* xq = (unsigned char*)d_ws;
  unsigned char* wq = xq + (size_t)M_ROWS * K_DIM;
  float4* pX = (float4*)(wq + (size_t)N_ROWS * K_DIM);
  float4* pW = pX + M_ROWS;

  quant_x_kernel<<<M_ROWS, 256, 0, stream>>>(x, xq, pX);
  quant_w_kernel<<<N_ROWS, 256, 0, stream>>>(wgt, bias, wq, pW);

  dim3 grid(N_ROWS / 128, M_ROWS / 128);
  wmma_gemm_kernel<<<grid, 256, 0, stream>>>(xq, wq, pX, pW, out);
}
